// DeformableTransformer_19370302505600
// MI455X (gfx1250) — matrix operand order repaired and checked by reading, not compiled, not run on hardware
//
#include <hip/hip_runtime.h>

// ---------------------------------------------------------------------------
// Multi-scale deformable attention for MI455X (gfx1250, wave32, WMMA).
//   N=2, Lq=Len_in=21760, C=256, M=8, L=4, P=4, D=32. ROWS = N*Lq = 43520.
// Pipeline:
//   1) value  = input_flatten @ W_val  + b_val      (WMMA bf16 GEMM, 256->256)
//   2) offs   = query         @ W_off  + b_off      (WMMA bf16 GEMM, 256->256)
//   3) attn   = query         @ W_attn + b_attn     (WMMA bf16 GEMM, 256->128)
//   4) softmax over 16 (L*P) per (row, head)
//   5) bilinear gather+accumulate: one wave per (row, head), lane = channel d
//   6) out    = acc @ W_out + b_out                 (WMMA bf16 GEMM, 256->256)
// ---------------------------------------------------------------------------

typedef __attribute__((ext_vector_type(16))) __bf16 v16bf;
typedef __attribute__((ext_vector_type(8)))  __bf16 v8bf;
typedef __attribute__((ext_vector_type(4)))  __bf16 v4bf;
typedef __attribute__((ext_vector_type(8)))  float  v8f;

#define BM 128
#define BN 64
#define BK 32
#define LDA 40   // bf16 elements; row stride 80B -> 16B aligned, 20-bank stride
#define LDB 40

// ---------------------------------------------------------------------------
// Tiled GEMM: C[M,Nc] = A[M,K] @ B[K,Nc] + bias[Nc]  (f32 in/out, bf16 WMMA)
// Grid: (Nc/BN, M/BM), 256 threads = 8 waves (2 over N, 4 over M).
// Each wave computes a 32x32 tile = 2x2 v_wmma_f32_16x16x32_bf16 fragments.
// ---------------------------------------------------------------------------
__launch_bounds__(256)
__global__ void gemm_bias_bf16_wmma(const float* __restrict__ A,
                                    const float* __restrict__ B,
                                    const float* __restrict__ bias,
                                    float* __restrict__ C,
                                    int Mrows, int Ncols, int K)
{
    __shared__ __bf16 ldsA[BM * LDA];
    __shared__ __bf16 ldsB[BN * LDB];

    const int t     = threadIdx.x;
    const int lane  = t & 31;
    const int wid   = t >> 5;
    const int widN  = wid & 1;      // 2 waves across N
    const int widM  = wid >> 1;     // 4 waves across M
    const int lhalf = lane >> 4;    // 0 or 1 (ISA half-wave select)
    const int lmod  = lane & 15;

    const int mBase = blockIdx.y * BM;
    const int nBase = blockIdx.x * BN;

    v8f c00 = {}, c01 = {}, c10 = {}, c11 = {};

    for (int kt = 0; kt < K; kt += BK) {
        __syncthreads();
        // ---- stage A tile: BM x BK f32 -> bf16 LDS (row-major, padded) ----
        #pragma unroll
        for (int i = 0; i < 4; ++i) {
            const int v   = t + i * 256;       // 1024 float4 chunks
            const int row = v >> 3;
            const int col = (v & 7) << 2;
            const float4 f = *(const float4*)(A + (size_t)(mBase + row) * K + kt + col);
            v4bf h;
            h[0] = (__bf16)f.x; h[1] = (__bf16)f.y;
            h[2] = (__bf16)f.z; h[3] = (__bf16)f.w;
            *(v4bf*)(&ldsA[row * LDA + col]) = h;
        }
        // ---- stage B tile: BK x BN f32 -> bf16 LDS transposed [n][k] ----
        #pragma unroll
        for (int i = 0; i < 2; ++i) {
            const int v = t + i * 256;         // 512 float4 chunks
            const int k = v >> 4;
            const int n = (v & 15) << 2;
            const float4 f = *(const float4*)(B + (size_t)(kt + k) * Ncols + nBase + n);
            ldsB[(n + 0) * LDB + k] = (__bf16)f.x;
            ldsB[(n + 1) * LDB + k] = (__bf16)f.y;
            ldsB[(n + 2) * LDB + k] = (__bf16)f.z;
            ldsB[(n + 3) * LDB + k] = (__bf16)f.w;
        }
        __syncthreads();

        // ---- A fragments (16x32 bf16, ISA layout):
        //   lanes 0-15 : row M=lane,    K in {0..7, 16..23}
        //   lanes 16-31: row M=lane-16, K in {8..15, 24..31}
        auto loadA = [&](int rbase) -> v16bf {
            const __bf16* p = &ldsA[(rbase + lmod) * LDA + lhalf * 8];
            const v8bf lo = *(const v8bf*)(p);
            const v8bf hi = *(const v8bf*)(p + 16);
            v16bf r;
            #pragma unroll
            for (int i = 0; i < 8; ++i) { r[i] = lo[i]; r[i + 8] = hi[i]; }
            return r;
        };
        // ---- B fragments (32x16 bf16):
        //   lanes 0-15 : col N=lane,    K in {0..7, 16..23}
        //   lanes 16-31: col N=lane-16, K in {8..15, 24..31}
        auto loadB = [&](int nb) -> v16bf {
            const __bf16* p = &ldsB[(nb + lmod) * LDB + lhalf * 8];
            const v8bf lo = *(const v8bf*)(p);
            const v8bf hi = *(const v8bf*)(p + 16);
            v16bf r;
            #pragma unroll
            for (int i = 0; i < 8; ++i) { r[i] = lo[i]; r[i + 8] = hi[i]; }
            return r;
        };

        const v16bf a0 = loadA(widM * 32);
        const v16bf a1 = loadA(widM * 32 + 16);
        const v16bf b0 = loadB(widN * 32);
        const v16bf b1 = loadB(widN * 32 + 16);

        c00 = __builtin_amdgcn_wmma_f32_16x16x32_bf16(false, a0, false, b0, (short)0, c00, false, false);
        c01 = __builtin_amdgcn_wmma_f32_16x16x32_bf16(false, a0, false, b1, (short)0, c01, false, false);
        c10 = __builtin_amdgcn_wmma_f32_16x16x32_bf16(false, a1, false, b0, (short)0, c10, false, false);
        c11 = __builtin_amdgcn_wmma_f32_16x16x32_bf16(false, a1, false, b1, (short)0, c11, false, false);
    }

    // ---- epilogue: C/D layout -> VGPR r holds row (r + lhalf*8), col lmod ----
    const int rowBase = mBase + widM * 32 + lhalf * 8;
    const int colBase = nBase + widN * 32 + lmod;
    const float bias0 = bias[colBase];
    const float bias1 = bias[colBase + 16];
    #pragma unroll
    for (int r = 0; r < 8; ++r) {
        C[(size_t)(rowBase + r)      * Ncols + colBase]      = c00[r] + bias0;
        C[(size_t)(rowBase + r)      * Ncols + colBase + 16] = c01[r] + bias1;
        C[(size_t)(rowBase + 16 + r) * Ncols + colBase]      = c10[r] + bias0;
        C[(size_t)(rowBase + 16 + r) * Ncols + colBase + 16] = c11[r] + bias1;
    }
}

// ---------------------------------------------------------------------------
// Softmax over 16 contiguous logits per (row, head). One thread per group.
// ---------------------------------------------------------------------------
__launch_bounds__(256)
__global__ void softmax16_kernel(float* __restrict__ attn, int ngroups)
{
    const int tid = blockIdx.x * 256 + threadIdx.x;
    if (tid >= ngroups) return;
    float* p = attn + (size_t)tid * 16;
    float v[16];
    float mx = -3.4e38f;
    #pragma unroll
    for (int i = 0; i < 16; ++i) { v[i] = p[i]; mx = fmaxf(mx, v[i]); }
    float s = 0.f;
    #pragma unroll
    for (int i = 0; i < 16; ++i) { v[i] = __expf(v[i] - mx); s += v[i]; }
    const float inv = 1.0f / s;
    #pragma unroll
    for (int i = 0; i < 16; ++i) p[i] = v[i] * inv;
}

// ---------------------------------------------------------------------------
// Sampling: one wave per (row = n*Lq+q, head m); lane = channel d (D==32).
// Gathers are 128B coalesced per wave; value buffer (44.6MB) lives in L2.
// grid_sample semantics: bilinear, padding_mode='zeros', align_corners=False.
// ---------------------------------------------------------------------------
__device__ __forceinline__ float msda_fetch(const float* __restrict__ base,
                                            int ix, int iy, int W, int H)
{
    const bool valid = ((unsigned)ix < (unsigned)W) & ((unsigned)iy < (unsigned)H);
    const int cx = ix < 0 ? 0 : (ix > W - 1 ? W - 1 : ix);
    const int cy = iy < 0 ? 0 : (iy > H - 1 ? H - 1 : iy);
    const float g = base[(size_t)(cy * W + cx) * 256];
    return valid ? g : 0.0f;
}

__launch_bounds__(256)
__global__ void msda_sample_kernel(const float* __restrict__ value,   // (N,Len,256)
                                   const float* __restrict__ offs,    // (rows,256)
                                   const float* __restrict__ attn,    // (rows,128)
                                   const float* __restrict__ refp,    // (rows,4,2)
                                   float* __restrict__ out)           // (rows,256)
{
    const int lane = threadIdx.x & 31;
    const int wid  = threadIdx.x >> 5;
    const int w    = blockIdx.x * 8 + wid;    // 0 .. 348159
    const int m    = w & 7;
    const int row  = w >> 3;                  // n*Lq + q
    const int n    = row / 21760;

    const float* __restrict__ vbase = value + ((size_t)n * 21760) * 256 + m * 32 + lane;
    const float* __restrict__ offp  = offs + (size_t)row * 256 + m * 32;
    const float* __restrict__ attp  = attn + (size_t)row * 128 + m * 16;

    const int   HW[4]   = {128, 64, 32, 16};
    const int   STT[4]  = {0, 16384, 20480, 21504};

    float acc = 0.0f;
    #pragma unroll
    for (int l = 0; l < 4; ++l) {
        const int   Hl   = HW[l];
        const int   Wl   = HW[l];
        const float invW = 1.0f / (float)Wl;
        const float invH = 1.0f / (float)Hl;
        const float rx = refp[((size_t)row * 4 + l) * 2 + 0];
        const float ry = refp[((size_t)row * 4 + l) * 2 + 1];
        const float* __restrict__ vlvl = vbase + (size_t)STT[l] * 256;

        #pragma unroll
        for (int p = 0; p < 4; ++p) {
            const float ox = offp[l * 8 + p * 2 + 0];
            const float oy = offp[l * 8 + p * 2 + 1];
            // x = loc_x * W - 0.5 ; y = loc_y * H - 0.5  (align_corners=False)
            const float x = (rx + ox * invW) * (float)Wl - 0.5f;
            const float y = (ry + oy * invH) * (float)Hl - 0.5f;
            const float x0f = floorf(x), y0f = floorf(y);
            const float fx = x - x0f, fy = y - y0f;
            const int x0 = (int)x0f, y0 = (int)y0f;
            const float wgt = attp[l * 4 + p];

            const float g00 = msda_fetch(vlvl, x0,     y0,     Wl, Hl);
            const float g01 = msda_fetch(vlvl, x0 + 1, y0,     Wl, Hl);
            const float g10 = msda_fetch(vlvl, x0,     y0 + 1, Wl, Hl);
            const float g11 = msda_fetch(vlvl, x0 + 1, y0 + 1, Wl, Hl);

            const float bil = g00 * (1.0f - fx) * (1.0f - fy)
                            + g01 * fx          * (1.0f - fy)
                            + g10 * (1.0f - fx) * fy
                            + g11 * fx          * fy;
            acc += wgt * bil;
        }
    }
    out[(size_t)row * 256 + m * 32 + lane] = acc;
}

// ---------------------------------------------------------------------------
// Launch. Inputs (setup_inputs order):
//  0 query (N,Lq,256) 1 reference_points (N,Lq,4,2) 2 input_flatten (N,Len,256)
//  3 W_off (256,256)  4 b_off (256)  5 W_attn (256,128) 6 b_attn (128)
//  7 W_val (256,256)  8 b_val (256)  9 W_out (256,256) 10 b_out (256)
// ---------------------------------------------------------------------------
extern "C" void kernel_launch(void* const* d_in, const int* in_sizes, int n_in,
                              void* d_out, int out_size, void* d_ws, size_t ws_size,
                              hipStream_t stream)
{
    const float* query  = (const float*)d_in[0];
    const float* refp   = (const float*)d_in[1];
    const float* inpf   = (const float*)d_in[2];
    const float* W_off  = (const float*)d_in[3];
    const float* b_off  = (const float*)d_in[4];
    const float* W_attn = (const float*)d_in[5];
    const float* b_attn = (const float*)d_in[6];
    const float* W_val  = (const float*)d_in[7];
    const float* b_val  = (const float*)d_in[8];
    const float* W_out  = (const float*)d_in[9];
    const float* b_out  = (const float*)d_in[10];
    float* out = (float*)d_out;
    float* ws  = (float*)d_ws;

    const size_t ROWS = 43520;                    // N * Lq
    float* value = ws;                            // ROWS*256
    float* offs  = ws + ROWS * 256;               // ROWS*256
    float* attn  = ws + ROWS * 512;               // ROWS*128
    float* acc   = ws + ROWS * 512 + ROWS * 128;  // ROWS*256

    const dim3 blk(256);
    // 43520 / 128 = 340 M-tiles exactly; K=256 = 8*BK exactly.
    gemm_bias_bf16_wmma<<<dim3(4, 340), blk, 0, stream>>>(inpf,  W_val,  b_val,  value, 43520, 256, 256);
    gemm_bias_bf16_wmma<<<dim3(4, 340), blk, 0, stream>>>(query, W_off,  b_off,  offs,  43520, 256, 256);
    gemm_bias_bf16_wmma<<<dim3(2, 340), blk, 0, stream>>>(query, W_attn, b_attn, attn,  43520, 128, 256);
    softmax16_kernel<<<1360, blk, 0, stream>>>(attn, 348160);
    msda_sample_kernel<<<43520, blk, 0, stream>>>(value, offs, attn, refp, acc);
    gemm_bias_bf16_wmma<<<dim3(4, 340), blk, 0, stream>>>(acc, W_out, b_out, out, 43520, 256, 256);
}
